// customVGG2_34067680592005
// MI455X (gfx1250) — matrix operand order repaired and checked by reading, not compiled, not run on hardware
//
#include <hip/hip_runtime.h>
#include <hip/hip_bf16.h>
#include <math.h>
#include <stdint.h>

typedef __attribute__((ext_vector_type(16))) _Float16 v16h;
typedef __attribute__((ext_vector_type(8)))  float    v8f;
typedef __attribute__((ext_vector_type(4)))  unsigned int u32x4;
typedef __attribute__((ext_vector_type(8)))  int      i32x8;
typedef __attribute__((ext_vector_type(4)))  int      i32x4;

#define BN_EPS 1e-5f
#define KNN_K  29

// ---------------- TDM helper: 2D tile (64 rows x 128 f32) -> LDS ----------------
// D# per CDNA5 ISA ch.8: group0 = {count/lds_addr/global_addr/type}, group1 =
// {flags, tensor dims, tile dims, stride}. pad: 1 dword after every 128 dwords
// -> LDS row stride 129 floats (conflict-free row-indexed reads).
__device__ static inline void tdm_load_2d(unsigned lds_off, const void* gptr,
                                          unsigned rows_valid) {
    unsigned long long ga = (unsigned long long)(uintptr_t)gptr;
    u32x4 g0;
    g0.x = 1u;                                             // count=1, user D#
    g0.y = lds_off;                                        // LDS byte address
    g0.z = (unsigned)(ga & 0xFFFFFFFFull);                 // global_addr[31:0]
    g0.w = (unsigned)((ga >> 32) & 0x01FFFFFFull)          // global_addr[56:32]
         | 0x80000000u;                                    // type=2 ("image")
    i32x8 g1;
    g1[0] = (int)((2u << 16) | (1u << 20) | (6u << 22));   // 4B elems, pad_en, 1dw/128dw
    g1[1] = (int)((8192u & 0xFFFFu) << 16);                // tensor_dim0 lo (8192)
    g1[2] = (int)((8192u >> 16) | ((rows_valid & 0xFFFFu) << 16)); // dim0 hi | tensor_dim1 lo
    g1[3] = (int)((rows_valid >> 16) | (128u << 16));      // tensor_dim1 hi | tile_dim0=128
    g1[4] = 64;                                            // tile_dim1=64, tile_dim2=0
    g1[5] = 8192;                                          // tensor_dim0_stride lo
    g1[6] = 0;                                             // stride hi | dim1_stride lo
    g1[7] = 0;
    i32x4 z4 = {0, 0, 0, 0};                               // 2D: groups 2/3 unused
    i32x8 z8 = {0, 0, 0, 0, 0, 0, 0, 0};
    __builtin_amdgcn_tensor_load_to_lds(g0, g1, z4, z4, z8, 0);
}

// ---------------- Kernel 0: repack conv2 weights to f16, tap-major ----------------
// out[o][t*64 + ci] = w2[o][ci][t]  (t = dy*3+dx). Tap-major k keeps each WMMA
// 32-chunk inside one tap -> branch-free im2col addressing in conv2.
__global__ void prep_w2h_kernel(const float* __restrict__ w2, _Float16* __restrict__ w2h) {
    int i = blockIdx.x * blockDim.x + threadIdx.x;
    if (i >= 128 * 576) return;
    int o = i / 576, r = i - o * 576;
    int t = r >> 6, ci = r & 63;
    w2h[i] = (_Float16)w2[o * 576 + ci * 9 + t];
}

// ---------------- Kernel 1: conv1 + BN + ReLU + maxpool2 ----------------
__global__ void conv1_pool_kernel(const float* __restrict__ imgs,
                                  const float* __restrict__ w,
                                  const float* __restrict__ cb,
                                  const float* __restrict__ sc,
                                  const float* __restrict__ bi,
                                  const float* __restrict__ mu,
                                  const float* __restrict__ var,
                                  float* __restrict__ p1) {
    int idx = blockIdx.x * blockDim.x + threadIdx.x;
    if (idx >= 64 * 64 * 16 * 16) return;
    int px = idx & 15, py = (idx >> 4) & 15, o = (idx >> 8) & 63, b = idx >> 14;

    float wr[27];
    #pragma unroll
    for (int j = 0; j < 27; ++j) wr[j] = w[o * 27 + j];
    float alpha = sc[o] * rsqrtf(var[o] + BN_EPS);
    float beta  = bi[o] + alpha * (cb[o] - mu[o]);

    float m = 0.0f;
    #pragma unroll
    for (int sy = 0; sy < 2; ++sy)
    #pragma unroll
    for (int sx = 0; sx < 2; ++sx) {
        int y = 2 * py + sy, x = 2 * px + sx;
        float acc = 0.f;
        #pragma unroll
        for (int ci = 0; ci < 3; ++ci)
        #pragma unroll
        for (int dy = 0; dy < 3; ++dy) {
            int iy = y + dy - 1;
            if ((unsigned)iy < 32u) {
                #pragma unroll
                for (int dx = 0; dx < 3; ++dx) {
                    int ix = x + dx - 1;
                    if ((unsigned)ix < 32u)
                        acc = fmaf(imgs[((b * 3 + ci) * 32 + iy) * 32 + ix],
                                   wr[ci * 9 + dy * 3 + dx], acc);
                }
            }
        }
        float v = fmaf(alpha, acc, beta);
        v = v > 0.f ? v : 0.f;
        m = v > m ? v : m;
    }
    p1[idx] = m;
}

// ---------------- Kernel 2: conv2 as WMMA implicit GEMM + BN + ReLU ----------------
// k = tap*64 + ci (tap-major). Each chunk of 32 k's has uniform (dy,dx) and a
// contiguous ci range -> multiply-free A-tile addressing, packed b32 LDS stores.
__global__ void __launch_bounds__(256) conv2_wmma_kernel(
        const float* __restrict__ p1,
        const _Float16* __restrict__ w2h,  // [128][576] tap-major
        const float* __restrict__ cb, const float* __restrict__ sc,
        const float* __restrict__ bi, const float* __restrict__ mu,
        const float* __restrict__ var,
        float* __restrict__ h2) {
    __shared__ _Float16 As[16][34];

    const int tid  = threadIdx.x;
    const int lane = tid & 31;
    const int wv   = tid >> 5;
    const int n0   = wv * 16;
    const int m0   = blockIdx.x * 16;
    const int b    = m0 >> 8;
    const int y    = (m0 >> 4) & 15;

    const int Mloc = lane & 15;
    const int hi   = lane >> 4;
    const int ncol = n0 + Mloc;

    // loader mapping: 256 threads -> 16 rows x 16 ci-pairs
    const int lrow = tid >> 4;
    const int lkk  = (tid & 15) * 2;

    v8f acc = {};

    #pragma unroll
    for (int k0 = 0; k0 < 576; k0 += 32) {
        const int tap = k0 >> 6;                 // constant after unroll
        const int ci0 = k0 & 63;
        const int dy  = tap / 3, dx = tap % 3;
        const int iy  = y + dy - 1;              // wave-uniform

        __syncthreads();
        {
            int ci = ci0 + lkk;
            int ix = lrow + dx - 1;
            float v0 = 0.f, v1 = 0.f;
            if ((unsigned)iy < 16u && (unsigned)ix < 16u) {
                const float* src = &p1[((b * 64 + ci) * 16 + iy) * 16 + ix];
                v0 = src[0];
                v1 = src[256];                   // next ci => +16*16
            }
            union { _Float16 h[2]; unsigned u; } pk;
            pk.h[0] = (_Float16)v0; pk.h[1] = (_Float16)v1;
            *(unsigned*)&As[lrow][lkk] = pk.u;
        }
        __syncthreads();

        // A frag: lanes 0-15 K=0..15, lanes 16-31 K=16..31 (contiguous halves)
        union { unsigned u[8]; v16h h; } af;
        const unsigned* rowp = (const unsigned*)&As[Mloc][0];
        int ub = hi * 8;
        #pragma unroll
        for (int j = 0; j < 4; ++j) {
            af.u[j]     = rowp[ub + j];
            af.u[4 + j] = rowp[ub + 4 + j];
        }
        // B frag: lanes 0-15 K=0..15, lanes 16-31 K=16..31 (contiguous halves)
        union { unsigned u[8]; v16h h; } bf;
        const unsigned* wp = (const unsigned*)(w2h + (size_t)ncol * 576 + k0 + hi * 16);
        #pragma unroll
        for (int j = 0; j < 8; ++j) bf.u[j] = wp[j];

        acc = __builtin_amdgcn_wmma_f32_16x16x32_f16(
                  false, af.h, false, bf.h, (short)0, acc, false, false);
    }

    float alpha = sc[ncol] * rsqrtf(var[ncol] + BN_EPS);
    float beta  = bi[ncol] + alpha * (cb[ncol] - mu[ncol]);
    #pragma unroll
    for (int r = 0; r < 8; ++r) {
        int M = r + hi * 8;
        float v = fmaf(alpha, acc[r], beta);
        v = v > 0.f ? v : 0.f;
        h2[(size_t)(m0 + M) * 128 + ncol] = v;
    }
}

// ---------------- Kernel 3: maxpool2 -> feats [64][8192] ----------------
__global__ void pool2_kernel(const float* __restrict__ h2, float* __restrict__ feats) {
    int idx = blockIdx.x * blockDim.x + threadIdx.x;
    if (idx >= 64 * 128 * 8 * 8) return;
    int px = idx & 7, py = (idx >> 3) & 7, o = (idx >> 6) & 127, b = idx >> 13;
    float m = 0.f;
    #pragma unroll
    for (int sy = 0; sy < 2; ++sy)
    #pragma unroll
    for (int sx = 0; sx < 2; ++sx) {
        int y = 2 * py + sy, x = 2 * px + sx;
        float v = h2[(size_t)(b * 256 + y * 16 + x) * 128 + o];
        m = v > m ? v : m;
    }
    feats[(size_t)b * 8192 + o * 64 + py * 8 + px] = m;
}

// ---------------- Kernel 4: L1 distances, TDM double-buffered ----------------
// Wave 0 issues tensor_load_to_lds for the next chunk (train + test tiles) while
// all 8 waves consume the current one; s_wait_tensorcnt(2) gates on the in-order
// completion of the current chunk's two loads. TDM pad gives 129-float row
// stride -> conflict-free LDS reads.
#define DC 128
#define RS 129
__global__ void __launch_bounds__(256) l1dist_kernel(
        const float* __restrict__ feats,   // [64][8192]
        const float* __restrict__ train,   // [10000][8192]
        float* __restrict__ Dpart) {       // [4][64][10000]
    __shared__ float sF[2][64 * RS];
    __shared__ float sT[2][64 * RS];

    const int tid = threadIdx.x;
    const int n0  = blockIdx.x * 64;
    const int seg = blockIdx.y;
    const int d_begin = seg * 2048;
    const unsigned rows = (unsigned)((10000 - n0) < 64 ? (10000 - n0) : 64);

    const int tx = tid & 15, ty = tid >> 4;

    float acc[4][4];
    #pragma unroll
    for (int i = 0; i < 4; ++i)
        #pragma unroll
        for (int j = 0; j < 4; ++j) acc[i][j] = 0.f;

    unsigned offF[2], offT[2];
    offF[0] = (unsigned)(uintptr_t)&sF[0][0];
    offF[1] = (unsigned)(uintptr_t)&sF[1][0];
    offT[0] = (unsigned)(uintptr_t)&sT[0][0];
    offT[1] = (unsigned)(uintptr_t)&sT[1][0];

    const float* tbase = train + (size_t)n0 * 8192 + d_begin;
    const float* fbase = feats + d_begin;

    if (tid < 32) {                               // wave 0 only (uniform branch)
        tdm_load_2d(offT[0], tbase, rows);
        tdm_load_2d(offF[0], fbase, 64u);
    }

    for (int c = 0; c < 16; ++c) {
        const int cur = c & 1, nxt = cur ^ 1;
        __syncthreads();                          // buf[nxt] free for reuse
        if (tid < 32) {
            if (c + 1 < 16) {
                tdm_load_2d(offT[nxt], tbase + (c + 1) * DC, rows);
                tdm_load_2d(offF[nxt], fbase + (c + 1) * DC, 64u);
                __builtin_amdgcn_s_wait_tensorcnt(2);   // chunk c's loads done
            } else {
                __builtin_amdgcn_s_wait_tensorcnt(0);
            }
        }
        __syncthreads();                          // data visible to all waves

        const float* fb = &sF[cur][0];
        const float* tb = &sT[cur][0];
        #pragma unroll 2
        for (int d = 0; d < DC; ++d) {
            float fa[4], tv[4];
            #pragma unroll
            for (int i = 0; i < 4; ++i) fa[i] = fb[(ty * 4 + i) * RS + d];
            #pragma unroll
            for (int j = 0; j < 4; ++j) tv[j] = tb[(tx * 4 + j) * RS + d];
            #pragma unroll
            for (int i = 0; i < 4; ++i)
                #pragma unroll
                for (int j = 0; j < 4; ++j)
                    acc[i][j] += fabsf(fa[i] - tv[j]);
        }
    }

    #pragma unroll
    for (int i = 0; i < 4; ++i) {
        int bb = ty * 4 + i;
        #pragma unroll
        for (int j = 0; j < 4; ++j) {
            int n = n0 + tx * 4 + j;
            if (n < 10000)
                Dpart[((size_t)seg * 64 + bb) * 10000 + n] = acc[i][j];
        }
    }
}

// ---------------- Kernel 5: top-29 + median + mode + weighted fraction ----------------
__global__ void __launch_bounds__(256) knn_kernel(
        const float* __restrict__ Dpart,
        const int*   __restrict__ labels,
        float* __restrict__ out) {
    const int b = blockIdx.x;
    const int tid = threadIdx.x;
    __shared__ float sv[256];
    __shared__ int   si[256];
    __shared__ int   so[256];
    __shared__ float kv[KNN_K];
    __shared__ int   ki[KNN_K];
    __shared__ int   winner;

    float lv[KNN_K]; int li[KNN_K];
    #pragma unroll
    for (int k = 0; k < KNN_K; ++k) { lv[k] = INFINITY; li[k] = 0x7fffffff; }

    const float* p0 = Dpart + (size_t)b * 10000;
    const size_t segstride = (size_t)64 * 10000;
    for (int n = tid; n < 10000; n += 256) {
        float d = p0[n] + p0[segstride + n] + p0[2 * segstride + n] + p0[3 * segstride + n];
        if (d < lv[KNN_K - 1]) {
            int p = KNN_K - 1;
            while (p > 0 && lv[p - 1] > d) { lv[p] = lv[p - 1]; li[p] = li[p - 1]; --p; }
            lv[p] = d; li[p] = n;
        }
    }

    int head = 0;
    for (int sel = 0; sel < KNN_K; ++sel) {
        sv[tid] = (head < KNN_K) ? lv[head] : INFINITY;
        si[tid] = (head < KNN_K) ? li[head] : 0x7fffffff;
        so[tid] = tid;
        __syncthreads();
        for (int s = 128; s > 0; s >>= 1) {
            if (tid < s) {
                float v1 = sv[tid], v2 = sv[tid + s];
                int   i1 = si[tid], i2 = si[tid + s];
                if (v2 < v1 || (v2 == v1 && i2 < i1)) {
                    sv[tid] = v2; si[tid] = i2; so[tid] = so[tid + s];
                }
            }
            __syncthreads();
        }
        if (tid == 0) { kv[sel] = sv[0]; ki[sel] = si[0]; winner = so[0]; }
        __syncthreads();
        if (tid == winner) ++head;
        __syncthreads();
    }

    if (tid == 0) {
        float s = kv[KNN_K / 2];
        int counts[10];
        #pragma unroll
        for (int c = 0; c < 10; ++c) counts[c] = 0;
        int lk[KNN_K];
        for (int j = 0; j < KNN_K; ++j) { lk[j] = labels[ki[j]]; counts[lk[j]]++; }
        int pred = 0;
        for (int c = 1; c < 10; ++c) if (counts[c] > counts[pred]) pred = c;
        float nr = 0.f, dr = 0.f;
        for (int j = 0; j < KNN_K; ++j) {
            float w = __expf(-kv[j] / s);
            dr += w;
            if (lk[j] == pred) nr += w;
        }
        float frac = nr / dr;
        out[b * 2 + 0] = frac;
        out[b * 2 + 1] = 1.f - frac;
    }
}

// ---------------- Host-side launch ----------------
extern "C" void kernel_launch(void* const* d_in, const int* in_sizes, int n_in,
                              void* d_out, int out_size, void* d_ws, size_t ws_size,
                              hipStream_t stream) {
    const float* imgs  = (const float*)d_in[0];
    const float* w1    = (const float*)d_in[1];
    const float* b1    = (const float*)d_in[2];
    const float* s1    = (const float*)d_in[3];
    const float* bi1   = (const float*)d_in[4];
    const float* m1    = (const float*)d_in[5];
    const float* v1    = (const float*)d_in[6];
    const float* w2    = (const float*)d_in[7];
    const float* b2    = (const float*)d_in[8];
    const float* s2    = (const float*)d_in[9];
    const float* bi2   = (const float*)d_in[10];
    const float* m2    = (const float*)d_in[11];
    const float* v2    = (const float*)d_in[12];
    const float* train = (const float*)d_in[13];
    const int*   labels= (const int*)d_in[14];
    float* out = (float*)d_out;

    char* ws = (char*)d_ws;
    float*    feats = (float*)(ws + 0);          //  2,097,152 B  [64][8192]
    float*    Dpart = (float*)(ws + 2097152);    // 10,240,000 B  [4][64][10000]
    float*    p1    = (float*)(ws + 12337152);   //  4,194,304 B  [64][64][16][16]
    float*    h2    = (float*)(ws + 16531456);   //  8,388,608 B  [16384][128]
    _Float16* w2h   = (_Float16*)(ws + 24920064);//    147,456 B  [128][576]

    prep_w2h_kernel  <<<288, 256, 0, stream>>>(w2, w2h);
    conv1_pool_kernel<<<4096, 256, 0, stream>>>(imgs, w1, b1, s1, bi1, m1, v1, p1);
    conv2_wmma_kernel<<<1024, 256, 0, stream>>>(p1, w2h, b2, s2, bi2, m2, v2, h2);
    pool2_kernel     <<<2048, 256, 0, stream>>>(h2, feats);
    dim3 gdist(157, 4);
    l1dist_kernel    <<<gdist, 256, 0, stream>>>(feats, train, Dpart);
    knn_kernel       <<<64, 256, 0, stream>>>(Dpart, labels, out);
}
